// Attention_75299366633572
// MI455X (gfx1250) — matrix operand order repaired and checked by reading, not compiled, not run on hardware
//
#include <hip/hip_runtime.h>
#include <hip/hip_bf16.h>

// ---------- types ----------
typedef __attribute__((ext_vector_type(16))) __bf16        v16bf;
typedef __attribute__((ext_vector_type(8)))  float         v8f;
typedef __attribute__((ext_vector_type(16))) unsigned short u16x16;
typedef __attribute__((ext_vector_type(8)))  unsigned short u16x8;
typedef __attribute__((ext_vector_type(4)))  unsigned short u16x4;
typedef __attribute__((ext_vector_type(4)))  float         f32x4;

#define DMODEL 128
#define NB     128
#define SEQ    1024
#define MTOT   (SEQ * DMODEL)   // 131072
#define F3     384
#define SPLITK 256              // k2 blocks

// float -> bf16, round-to-nearest-even
__device__ __forceinline__ unsigned short f2bf(float f) {
  unsigned u = __float_as_uint(f);
  u += 0x7FFFu + ((u >> 16) & 1u);
  return (unsigned short)(u >> 16);
}

// A-fragment (16x32 bf16, row-major in LDS, stride lda elems)
// lanes 0-15: M=lane, K {0..7}+{16..23}; lanes 16-31: M=lane-16, K {8..15}+{24..31}
__device__ __forceinline__ u16x16 frag_a_lds(const unsigned short* base, int lda) {
  const int lane = threadIdx.x & 31;
  const int row  = lane & 15;
  const int koff = (lane >> 4) << 3;            // 0 or 8
  const unsigned short* p = base + row * lda + koff;
  u16x8 lo = *(const u16x8*)(p);                // ds_load_b128
  u16x8 hi = *(const u16x8*)(p + 16);
  return __builtin_shufflevector(lo, hi, 0,1,2,3,4,5,6,7,8,9,10,11,12,13,14,15);
}

// B-fragment (32x16 bf16). Storage is [N][K] row-major (row = output column,
// contiguous along K), stride ldb elems. lanes 0-15: K {0..7}+{16..23}; lanes 16-31: K {8..15}+{24..31}.
__device__ __forceinline__ u16x16 frag_b_lds(const unsigned short* base, int ldb) {
  const int lane = threadIdx.x & 31;
  const int col  = lane & 15;
  const int koff = (lane >> 4) << 3;            // 0 or 8
  const unsigned short* p = base + col * ldb + koff;
  u16x8 lo = *(const u16x8*)(p);
  u16x8 hi = *(const u16x8*)(p + 16);
  return __builtin_shufflevector(lo, hi, 0,1,2,3,4,5,6,7,8,9,10,11,12,13,14,15);
}

__device__ __forceinline__ v8f wmma_bf16(u16x16 a, u16x16 b, v8f c) {
  return __builtin_amdgcn_wmma_f32_16x16x32_bf16(
      false, __builtin_bit_cast(v16bf, a),
      false, __builtin_bit_cast(v16bf, b),
      (short)0, c, false, false);
}

// ---------- k0: de-interleave W rows (q|k|v) and convert to bf16 ----------
__global__ __launch_bounds__(128) void k0_prep(const float* __restrict__ W,
                                               const float* __restrict__ b,
                                               unsigned short* __restrict__ Wp,
                                               float* __restrict__ bp) {
  const int fp = blockIdx.x;            // permuted feature 0..383
  const int which = fp >> 7;            // 0=q,1=k,2=v
  const int c = fp & 127;
  const int forig = 3 * c + which;
  const int d = threadIdx.x;
  Wp[fp * DMODEL + d] = f2bf(W[forig * DMODEL + d]);
  if (d == 0) bp[fp] = b[forig];
}

// ---------- k1: qkv projection (64 rows x 384 features per block) ----------
__global__ __launch_bounds__(256) void k1_qkv(const float* __restrict__ x,
                                              const unsigned short* __restrict__ Wp,
                                              const float* __restrict__ bp,
                                              unsigned short* __restrict__ Q,
                                              unsigned short* __restrict__ Km,
                                              unsigned short* __restrict__ V) {
  extern __shared__ char smem[];
  unsigned short* Wl = (unsigned short*)smem;            // 384*128 bf16 (96 KB)
  unsigned short* Al = Wl + F3 * DMODEL;                 // 64*128 bf16 (16 KB)
  float* bl = (float*)(Al + 64 * DMODEL);                // 384
  float* rs = bl + F3;                                   // 64 row scales

  const int t  = threadIdx.x;
  const int r0 = blockIdx.x * 64;        // flat (n,s) row
  const int n  = r0 >> 10;               // 64 | 1024 -> single n per block
  const int s0 = r0 & 1023;

  if (t < 64) {                          // temporal PE scale per row (depends on s only)
    int s = s0 + t;
    int p0 = s >> 5, p1 = s & 31;
    float arg = 3.14159265358979323846f * (float)p0 *
                __powf(1000.0f, -(float)(p1 & ~1) / 128.0f);
    rs[t] = (p1 & 1) ? __cosf(arg) : __sinf(arg);
  }
  if (t < 192) { bl[t] = bp[t]; bl[t + 192] = bp[t + 192]; }
  {
    const uint4* src = (const uint4*)Wp;                 // 8 bf16 per uint4
    uint4* dst = (uint4*)Wl;
    for (int i = t; i < (F3 * DMODEL) / 8; i += 256) dst[i] = src[i];
  }
  __syncthreads();
  {
    const f32x4* xs = (const f32x4*)(x + (size_t)r0 * DMODEL);
    for (int it = 0; it < 8; ++it) {
      int ch  = t + it * 256;            // 0..2047
      int row = ch >> 5;
      int c4  = (ch & 31) << 2;
      f32x4 vv = xs[ch];
      float sc = rs[row];
      u16x4 pk = { f2bf(vv.x * sc), f2bf(vv.y * sc), f2bf(vv.z * sc), f2bf(vv.w * sc) };
      *(u16x4*)(Al + row * DMODEL + c4) = pk;
    }
  }
  __syncthreads();

  const int w = t >> 5, rw = w & 3, cw = w >> 2;
  const int lane = t & 31, nIdx = lane & 15, mhi = (lane >> 4) << 3;
  v8f acc[12];
  for (int nt = 0; nt < 12; ++nt) {
    float bv = bl[cw * 192 + nt * 16 + nIdx];            // bias is an N-column term
    v8f z; for (int i = 0; i < 8; ++i) z[i] = bv;
    acc[nt] = z;
  }
  for (int kk = 0; kk < DMODEL; kk += 32) {
    u16x16 a = frag_a_lds(Al + rw * 16 * DMODEL + kk, DMODEL);
    for (int nt = 0; nt < 12; ++nt) {
      u16x16 bfr = frag_b_lds(Wl + (cw * 192 + nt * 16) * DMODEL + kk, DMODEL);
      acc[nt] = wmma_bf16(a, bfr, acc[nt]);
    }
  }
  for (int nt = 0; nt < 12; ++nt) {
    int fp = cw * 192 + nt * 16 + nIdx;
    int which = fp >> 7;
    int c = fp & 127;
    unsigned short* dst = which == 0 ? Q : (which == 1 ? Km : V);
    for (int i = 0; i < 8; ++i) {
      int srow = s0 + rw * 16 + mhi + i;
      dst[(size_t)n * MTOT + (size_t)srow * DMODEL + c] = f2bf(acc[nt][i]);
    }
  }
}

// ---------- k2: split-K Q*K^T partials (256 blocks x 512 k each) ----------
__global__ __launch_bounds__(256) void k2_scores(const unsigned short* __restrict__ Q,
                                                 const unsigned short* __restrict__ Km,
                                                 float* __restrict__ part) {
  extern __shared__ char smem[];
  unsigned short* Ql = (unsigned short*)smem;            // 128*256 bf16 (64 KB)
  unsigned short* Kl = Ql + 128 * 256;                   // 128*256 bf16 (64 KB)
  const int t = threadIdx.x, w = t >> 5, lane = t & 31, nIdx = lane & 15;
  const size_t mBase = (size_t)blockIdx.x * 512;

  v8f acc[8];
  for (int j = 0; j < 8; ++j) { v8f z; for (int i = 0; i < 8; ++i) z[i] = 0.f; acc[j] = z; }

  for (int mc = 0; mc < 512; mc += 256) {
    for (int it = 0; it < 16; ++it) {
      int ch = t + it * 256;             // 128 rows x 32 uint4
      int row = ch >> 5, c8 = (ch & 31) << 3;
      ((uint4*)Ql)[row * 32 + (ch & 31)] =
          *(const uint4*)(Q + (size_t)row * MTOT + mBase + mc + c8);
      ((uint4*)Kl)[row * 32 + (ch & 31)] =
          *(const uint4*)(Km + (size_t)row * MTOT + mBase + mc + c8);
    }
    __syncthreads();
    for (int kk = 0; kk < 256; kk += 32) {
      u16x16 a = frag_a_lds(Ql + w * 16 * 256 + kk, 256);
      for (int jt = 0; jt < 8; ++jt) {
        u16x16 b = frag_b_lds(Kl + jt * 16 * 256 + kk, 256);
        acc[jt] = wmma_bf16(a, b, acc[jt]);
      }
    }
    __syncthreads();
  }
  const int mhi = (lane >> 4) << 3;
  float* pb = part + (size_t)blockIdx.x * (128 * 128);
  for (int jt = 0; jt < 8; ++jt)
    for (int i = 0; i < 8; ++i)
      pb[(w * 16 + mhi + i) * 128 + jt * 16 + nIdx] = acc[jt][i];
}

// ---------- k3: deterministic partial reduction + row softmax ----------
__global__ __launch_bounds__(128) void k3_softmax(const float* __restrict__ part,
                                                  unsigned short* __restrict__ attn) {
  __shared__ float red[128];
  const int i = blockIdx.x, j = threadIdx.x;
  float sum = 0.f;
  for (int b = 0; b < SPLITK; ++b) sum += part[(size_t)b * 16384 + i * 128 + j];
  float sc = sum * 0.08838834764831845f;   // 1/sqrt(128)
  red[j] = sc; __syncthreads();
  for (int off = 64; off > 0; off >>= 1) {
    if (j < off) red[j] = fmaxf(red[j], red[j + off]);
    __syncthreads();
  }
  float mx = red[0]; __syncthreads();
  float e = __expf(sc - mx);
  red[j] = e; __syncthreads();
  for (int off = 64; off > 0; off >>= 1) {
    if (j < off) red[j] += red[j + off];
    __syncthreads();
  }
  attn[i * 128 + j] = f2bf(e / red[0]);
}

// ---------- k4: out = attn @ V (256 m-columns per block) ----------
#define VT_LD 136   // 128 + 8 pad; 272 B row stride keeps 16B-aligned b128 loads
__global__ __launch_bounds__(256) void k4_out(const unsigned short* __restrict__ attn,
                                              const unsigned short* __restrict__ V,
                                              float* __restrict__ out) {
  extern __shared__ char smem[];
  unsigned short* Al = (unsigned short*)smem;            // 128*128 bf16 (32 KB)
  unsigned short* vT = Al + 128 * 128;                   // 256*VT_LD bf16 (~68 KB)
  const int t = threadIdx.x;
  const size_t mBase = (size_t)blockIdx.x * 256;

  for (int it = 0; it < 8; ++it) {
    int ch = t + it * 256;
    ((uint4*)Al)[ch] = ((const uint4*)attn)[ch];
  }
  for (int it = 0; it < 16; ++it) {
    int ch = t + it * 256;               // 128 j-rows x 32 chunks of 8 m
    int j = ch >> 5, m8 = (ch & 31) << 3;
    u16x8 vv = *(const u16x8*)(V + (size_t)j * MTOT + mBase + m8);
    for (int e = 0; e < 8; ++e) vT[(m8 + e) * VT_LD + j] = vv[e];
  }
  __syncthreads();

  const int w = t >> 5, lane = t & 31, nIdx = lane & 15, mhi = (lane >> 4) << 3;
  v8f acc[16];
  for (int mt = 0; mt < 16; ++mt) { v8f z; for (int i = 0; i < 8; ++i) z[i] = 0.f; acc[mt] = z; }
  for (int kk = 0; kk < 128; kk += 32) {
    u16x16 a = frag_a_lds(Al + w * 16 * 128 + kk, 128);
    for (int mt = 0; mt < 16; ++mt) {
      u16x16 b = frag_b_lds(vT + mt * 16 * VT_LD + kk, VT_LD);
      acc[mt] = wmma_bf16(a, b, acc[mt]);
    }
  }
  for (int mt = 0; mt < 16; ++mt)
    for (int i = 0; i < 8; ++i)
      out[(size_t)(w * 16 + mhi + i) * MTOT + mBase + mt * 16 + nIdx] = acc[mt][i];
}

// ---------- launch ----------
extern "C" void kernel_launch(void* const* d_in, const int* in_sizes, int n_in,
                              void* d_out, int out_size, void* d_ws, size_t ws_size,
                              hipStream_t stream) {
  const float* x = (const float*)d_in[0];
  const float* W = (const float*)d_in[1];
  const float* b = (const float*)d_in[2];
  float* out = (float*)d_out;

  char* ws = (char*)d_ws;
  const size_t QB = (size_t)NB * MTOT * sizeof(unsigned short);   // 33.5 MB each
  unsigned short* Q    = (unsigned short*)(ws);
  unsigned short* Km   = (unsigned short*)(ws + QB);
  unsigned short* V    = (unsigned short*)(ws + 2 * QB);
  unsigned short* Wp   = (unsigned short*)(ws + 3 * QB);
  float*          bp   = (float*)(ws + 3 * QB + (size_t)F3 * DMODEL * 2);
  float*          part = (float*)(ws + 3 * QB + (size_t)F3 * DMODEL * 2 + 2048);
  unsigned short* attn = (unsigned short*)((char*)part + (size_t)SPLITK * 16384 * 4);

  hipLaunchKernelGGL(k0_prep, dim3(F3), dim3(128), 0, stream, W, b, Wp, bp);

  size_t s1 = (size_t)(F3 * DMODEL + 64 * DMODEL) * 2 + (F3 + 64) * 4;   // ~114 KB
  hipLaunchKernelGGL(k1_qkv, dim3(MTOT / 64), dim3(256), s1, stream, x, Wp, bp, Q, Km, V);

  size_t s2 = (size_t)2 * 128 * 256 * 2;                                 // 128 KB
  hipLaunchKernelGGL(k2_scores, dim3(SPLITK), dim3(256), s2, stream, Q, Km, part);

  hipLaunchKernelGGL(k3_softmax, dim3(128), dim3(128), 0, stream, part, attn);

  size_t s4 = (size_t)(128 * 128 + 256 * VT_LD) * 2;                     // ~100 KB
  hipLaunchKernelGGL(k4_out, dim3(MTOT / 256), dim3(256), s4, stream, attn, V, out);
}
